// SegTransformer_49478023250584
// MI455X (gfx1250) — compile-verified, matrix operand order checked by reading, not run on hardware
//
#include <hip/hip_runtime.h>
#include <math.h>

// ---------------------------------------------------------------------------
// Types for CDNA5 WMMA (wave32, 16x16x32 f16 -> f32)
// ---------------------------------------------------------------------------
typedef _Float16 f16_t;
typedef __attribute__((ext_vector_type(16))) _Float16 v16h;
typedef __attribute__((ext_vector_type(8)))  float    v8f;
typedef __attribute__((ext_vector_type(8)))  _Float16 h8v;
typedef __attribute__((ext_vector_type(4)))  float    f4v;

union HFrag { v16h v; f16_t e[16]; };
union FFrag { v8f  v; float e[8];  };

// ---------------------------------------------------------------------------
// Model dimensions
// ---------------------------------------------------------------------------
#define BB   8
#define NP   2048
#define GG   128
#define KG   32
#define DD   384
#define NH   6
#define DH   64
#define TT   128          // tokens per batch == GG
#define BT   (BB*TT)      // 1024
#define BGK  (BB*GG*KG)   // 32768
#define BN   (BB*NP)      // 16384
#define XCATC 1152
#define CLS  50
#define PKIN 1184         // prop c1 K padded (1155 -> 1184)

static inline int cdiv(long a, long b) { return (int)((a + b - 1) / b); }

// ---------------------------------------------------------------------------
// Generic batched WMMA GEMM:
//   C[z][M,N] = epi( alpha * A[z][M,K] * Wt[z]^T ) (+bias)(BN)(act)(+resid)
// A fp32 [M,K] (row stride lda); Wt f16 packed [N,K] (row stride ldw,
// contiguous along K). z = zb*H + zh with separate strides.
// Wave tile: 64(M) x 16(N), 4 accumulators sharing one B fragment.
// Block: 4 waves -> 64x64 tile. Fast path: unconditional 128-bit loads.
// ---------------------------------------------------------------------------
__global__ __launch_bounds__(128)
void gemm_wmma(const float* __restrict__ A, int lda, long sAb, long sAh,
               const f16_t* __restrict__ W, int ldw, long sWb, long sWh,
               const float* __restrict__ bias,
               const float* __restrict__ bng, const float* __restrict__ bnb,
               const float* __restrict__ bnm, const float* __restrict__ bnv,
               const float* __restrict__ resid,
               float* __restrict__ C, int ldc, long sCb, long sCh,
               int M, int N, int Kd, int H, float alpha, int act)
{
    int z  = blockIdx.z;
    int zb = z / H, zh = z - zb * H;
    A += zb * sAb + zh * sAh;
    W += zb * sWb + zh * sWh;
    C += zb * sCb + zh * sCh;
    if (resid) resid += zb * sCb + zh * sCh;

    const int lane  = threadIdx.x & 31;
    const int wv    = threadIdx.x >> 5;
    const int col0  = blockIdx.x * 64 + wv * 16;
    const int row0  = blockIdx.y * 64;
    if (col0 >= N) return;                       // wave-uniform exit

    const int rA0   = row0 + (lane & 15);
    const int cB    = col0 + (lane & 15);
    const int khalf = (lane >> 4) * 8;

    FFrag acc[4];
#pragma unroll
    for (int t = 0; t < 4; ++t)
#pragma unroll
        for (int r = 0; r < 8; ++r) acc[t].e[r] = 0.0f;

    const bool alignOK = ((lda & 3) == 0) && ((ldw & 7) == 0);
    const bool fullN   = (col0 + 16 <= N);
    const bool fullM   = (row0 + 64 <= M);

    for (int kb = 0; kb < Kd; kb += 32) {
        const bool fast = alignOK && fullN && fullM && (kb + 32 <= Kd);
        if (fast) {
            // ---- fast path: vectorized, branch-free ----
            const f16_t* wr = W + (long)cB * ldw + kb + khalf;
            h8v b0 = *(const h8v*)(wr);
            h8v b1 = *(const h8v*)(wr + 16);
            HFrag b;
#pragma unroll
            for (int j = 0; j < 8; ++j) { b.e[j] = b0[j]; b.e[8 + j] = b1[j]; }
            if (kb + 96 < Kd) __builtin_prefetch(wr + 64, 0, 1);
#pragma unroll
            for (int t = 0; t < 4; ++t) {
                const float* ar = A + (long)(rA0 + 16 * t) * lda + kb + khalf;
                f4v a0 = *(const f4v*)(ar);
                f4v a1 = *(const f4v*)(ar + 4);
                f4v a2 = *(const f4v*)(ar + 16);
                f4v a3 = *(const f4v*)(ar + 20);
                HFrag a;
#pragma unroll
                for (int j = 0; j < 4; ++j) {
                    a.e[j]      = (f16_t)a0[j];
                    a.e[4 + j]  = (f16_t)a1[j];
                    a.e[8 + j]  = (f16_t)a2[j];
                    a.e[12 + j] = (f16_t)a3[j];
                }
                acc[t].v = __builtin_amdgcn_wmma_f32_16x16x32_f16(
                    false, a.v, false, b.v, (short)0, acc[t].v, false, false);
            }
        } else {
            // ---- slow path: per-element guards (edges only) ----
            HFrag b;
#pragma unroll
            for (int i = 0; i < 16; ++i) {
                int kk = kb + khalf + i + ((i >= 8) ? 8 : 0);
                b.e[i] = (cB < N && kk < Kd) ? W[(long)cB * ldw + kk] : (f16_t)0.0f;
            }
#pragma unroll
            for (int t = 0; t < 4; ++t) {
                const int rA = rA0 + 16 * t;
                HFrag a;
#pragma unroll
                for (int i = 0; i < 16; ++i) {
                    int kk = kb + khalf + i + ((i >= 8) ? 8 : 0);
                    float av = (rA < M && kk < Kd) ? A[(long)rA * lda + kk] : 0.0f;
                    a.e[i] = (f16_t)av;
                }
                acc[t].v = __builtin_amdgcn_wmma_f32_16x16x32_f16(
                    false, a.v, false, b.v, (short)0, acc[t].v, false, false);
            }
        }
    }

    const int cn = cB;
    if (cn >= N) return;                          // after all WMMAs
    float bb = bias ? bias[cn] : 0.0f;
    float sc = 1.0f, sh = 0.0f;
    if (bng) { sc = bng[cn] * rsqrtf(bnv[cn] + 1e-5f); sh = bnb[cn] - bnm[cn] * sc; }
#pragma unroll
    for (int t = 0; t < 4; ++t) {
#pragma unroll
        for (int r = 0; r < 8; ++r) {
            int rm = row0 + 16 * t + r + (lane >> 4) * 8;
            if (rm >= M) continue;
            float v = acc[t].e[r] * alpha + bb;
            if (bng) v = v * sc + sh;
            if (act == 1) v = fmaxf(v, 0.0f);
            else if (act == 2) v = 0.5f * v * (1.0f + erff(v * 0.70710678f));
            if (resid) v += resid[(long)rm * ldc + cn];
            C[(long)rm * ldc + cn] = v;
        }
    }
}

// ---------------------------------------------------------------------------
// Packing kernels
// ---------------------------------------------------------------------------
__global__ void pack_f16_k(const float* __restrict__ in, f16_t* __restrict__ out, long n) {
    long i = (long)blockIdx.x * blockDim.x + threadIdx.x;
    if (i < n) out[i] = (f16_t)in[i];
}

// fp32 [K,N] row-major -> f16 [N,Kpad] (transposed, K zero-padded)
__global__ void packT_k(const float* __restrict__ in, f16_t* __restrict__ out,
                        int K, int N, int Kpad) {
    long i = (long)blockIdx.x * blockDim.x + threadIdx.x;
    if (i >= (long)N * Kpad) return;
    int n = (int)(i / Kpad), k = (int)(i % Kpad);
    out[i] = (k < K) ? (f16_t)in[(long)k * N + n] : (f16_t)0.0f;
}

// V slice of qkv -> f16 [B,H,Dh,T] (transposed for the att*V GEMM)
__global__ void pack_vT_k(const float* __restrict__ qkv, f16_t* __restrict__ vt) {
    long i = (long)blockIdx.x * blockDim.x + threadIdx.x;   // over B*NH*DH*TT
    if (i >= (long)BB * NH * DH * TT) return;
    int t = (int)(i % TT);
    long r = i / TT;
    int d = (int)(r % DH);
    long r2 = r / DH;
    int h = (int)(r2 % NH);
    int b = (int)(r2 / NH);
    vt[i] = (f16_t)qkv[((long)b * TT + t) * 1152 + 768 + h * 64 + d];
}

// ---------------------------------------------------------------------------
// Small utility kernels
// ---------------------------------------------------------------------------
__global__ void transpose_pts_k(const float* __restrict__ pts, float* __restrict__ xyz) {
    long i = (long)blockIdx.x * blockDim.x + threadIdx.x;   // over B*NP
    if (i >= (long)BB * NP) return;
    int b = (int)(i / NP), n = (int)(i % NP);
    for (int c = 0; c < 3; ++c)
        xyz[((long)b * NP + n) * 3 + c] = pts[(long)b * 3 * NP + (long)c * NP + n];
}

__global__ __launch_bounds__(256)
void fps_k(const float* __restrict__ xyz, int* __restrict__ fps_idx) {
    int b = blockIdx.x, tid = threadIdx.x;
    __shared__ float dist[NP];
    __shared__ float red[256];
    __shared__ int   redi[256];
    __shared__ int   cur;
    const float* p = xyz + (long)b * NP * 3;
    for (int i = tid; i < NP; i += 256) dist[i] = 1e10f;
    if (tid == 0) cur = 0;
    __syncthreads();
    for (int s = 0; s < GG; ++s) {
        int far = cur;
        if (tid == 0) fps_idx[b * GG + s] = far;
        float cx = p[far * 3], cy = p[far * 3 + 1], cz = p[far * 3 + 2];
        float best = -1.0f; int bi = 0;
        for (int i = tid; i < NP; i += 256) {
            float dx = p[i * 3] - cx, dy = p[i * 3 + 1] - cy, dz = p[i * 3 + 2] - cz;
            float d = dx * dx + dy * dy + dz * dz;
            float nd = fminf(dist[i], d);
            dist[i] = nd;
            if (nd > best) { best = nd; bi = i; }
        }
        red[tid] = best; redi[tid] = bi;
        __syncthreads();
        for (int o = 128; o > 0; o >>= 1) {
            if (tid < o && red[tid + o] > red[tid]) { red[tid] = red[tid + o]; redi[tid] = redi[tid + o]; }
            __syncthreads();
        }
        if (tid == 0) cur = redi[0];
        __syncthreads();
    }
}

__global__ void gather_center_k(const float* __restrict__ xyz, const int* __restrict__ fps_idx,
                                float* __restrict__ center) {
    long i = (long)blockIdx.x * blockDim.x + threadIdx.x;   // over B*G
    if (i >= (long)BB * GG) return;
    int b = (int)(i / GG);
    int idx = fps_idx[i];
    for (int c = 0; c < 3; ++c)
        center[i * 3 + c] = xyz[((long)b * NP + idx) * 3 + c];
}

__global__ __launch_bounds__(256)
void knn_k(const float* __restrict__ xyz, const float* __restrict__ center,
           int* __restrict__ nn_idx) {
    int bg = blockIdx.x, tid = threadIdx.x;
    int b = bg / GG;
    __shared__ float d2s[NP];
    __shared__ float red[256];
    __shared__ int   redi[256];
    float cx = center[bg * 3], cy = center[bg * 3 + 1], cz = center[bg * 3 + 2];
    const float* p = xyz + (long)b * NP * 3;
    for (int i = tid; i < NP; i += 256) {
        float dx = p[i * 3] - cx, dy = p[i * 3 + 1] - cy, dz = p[i * 3 + 2] - cz;
        d2s[i] = dx * dx + dy * dy + dz * dz;
    }
    __syncthreads();
    for (int k = 0; k < KG; ++k) {
        float best = 1e30f; int bi = 0;
        for (int i = tid; i < NP; i += 256)
            if (d2s[i] < best) { best = d2s[i]; bi = i; }
        red[tid] = best; redi[tid] = bi;
        __syncthreads();
        for (int o = 128; o > 0; o >>= 1) {
            if (tid < o && red[tid + o] < red[tid]) { red[tid] = red[tid + o]; redi[tid] = redi[tid + o]; }
            __syncthreads();
        }
        if (tid == 0) { nn_idx[(long)bg * KG + k] = redi[0]; d2s[redi[0]] = 1e30f; }
        __syncthreads();
    }
}

// writes nbr rows padded to 32 cols (cols 3..31 zero) for the K=32 GEMM
__global__ void build_nbr_k(const float* __restrict__ xyz, const float* __restrict__ center,
                            const int* __restrict__ nn_idx, float* __restrict__ nbr) {
    long i = (long)blockIdx.x * blockDim.x + threadIdx.x;   // over BGK
    if (i >= (long)BGK) return;
    int bg = (int)(i / KG);
    int b  = bg / GG;
    int idx = nn_idx[i];
    float* row = nbr + i * 32;
    for (int c = 0; c < 3; ++c)
        row[c] = xyz[((long)b * NP + idx) * 3 + c] - center[(long)bg * 3 + c];
    for (int c = 3; c < 32; ++c) row[c] = 0.0f;
}

__global__ void pool_max_k_k(const float* __restrict__ in, float* __restrict__ out,
                             int BGn, int Kk, int C) {
    long i = (long)blockIdx.x * blockDim.x + threadIdx.x;
    if (i >= (long)BGn * C) return;
    int bg = (int)(i / C), c = (int)(i % C);
    float m = -1e30f;
    for (int k = 0; k < Kk; ++k) m = fmaxf(m, in[((long)bg * Kk + k) * C + c]);
    out[(long)bg * C + c] = m;
}

__global__ void concat512_k(const float* __restrict__ g256, const float* __restrict__ f2,
                            float* __restrict__ out) {
    long i = (long)blockIdx.x * blockDim.x + threadIdx.x;   // over BGK*512
    if (i >= (long)BGK * 512) return;
    long row = i / 512; int c = (int)(i % 512);
    int bg = (int)(row / KG);
    out[i] = (c < 256) ? g256[(long)bg * 256 + c] : f2[row * 256 + (c - 256)];
}

__global__ void pos_embed_k(const float* __restrict__ center, float* __restrict__ pe) {
    long i = (long)blockIdx.x * blockDim.x + threadIdx.x;   // over BT*DD
    if (i >= (long)BT * DD) return;
    long bg = i / DD; int c = (int)(i % DD);
    int j = c / 128, t = c % 128;
    float coord = center[bg * 3 + j];
    float val;
    if (t < 64) {
        float f = expf(-logf(10000.0f) * (float)t / 64.0f);
        val = sinf(coord * f);
    } else {
        float f = expf(-logf(10000.0f) * (float)(t - 64) / 64.0f);
        val = cosf(coord * f);
    }
    pe[i] = val;
}

__global__ void add_vec_k(const float* __restrict__ a, const float* __restrict__ b,
                          float* __restrict__ c, long n) {
    long i = (long)blockIdx.x * blockDim.x + threadIdx.x;
    if (i < n) c[i] = a[i] + b[i];
}

__global__ __launch_bounds__(128)
void layernorm_k(const float* __restrict__ x, int ld_in,
                 const float* __restrict__ g, const float* __restrict__ b,
                 float* __restrict__ out, int ld_out, int out_off,
                 int rows, int D) {
    int row  = blockIdx.x * 4 + (threadIdx.x >> 5);
    int lane = threadIdx.x & 31;
    if (row >= rows) return;
    const float* xr = x + (long)row * ld_in;
    float s = 0.0f, ss = 0.0f;
    for (int c = lane; c < D; c += 32) { float v = xr[c]; s += v; ss += v * v; }
    for (int o = 16; o > 0; o >>= 1) { s += __shfl_xor(s, o, 32); ss += __shfl_xor(ss, o, 32); }
    float mean = s / D;
    float var  = ss / D - mean * mean;
    float rstd = rsqrtf(var + 1e-5f);
    float* orow = out + (long)row * ld_out + out_off;
    for (int c = lane; c < D; c += 32) orow[c] = (xr[c] - mean) * rstd * g[c] + b[c];
}

__global__ __launch_bounds__(128)
void softmax_rows_k(float* __restrict__ x, long rows, int D) {
    long row = (long)blockIdx.x * 4 + (threadIdx.x >> 5);
    int lane = threadIdx.x & 31;
    if (row >= rows) return;
    float* p = x + row * D;
    float mx = -1e30f;
    for (int c = lane; c < D; c += 32) mx = fmaxf(mx, p[c]);
    for (int o = 16; o > 0; o >>= 1) mx = fmaxf(mx, __shfl_xor(mx, o, 32));
    float s = 0.0f;
    for (int c = lane; c < D; c += 32) { float e = expf(p[c] - mx); p[c] = e; s += e; }
    for (int o = 16; o > 0; o >>= 1) s += __shfl_xor(s, o, 32);
    float inv = 1.0f / s;
    for (int c = lane; c < D; c += 32) p[c] *= inv;
}

__global__ void pool_global_k(const float* __restrict__ xcat, float* __restrict__ glob) {
    long i = (long)blockIdx.x * blockDim.x + threadIdx.x;   // over B*1152
    if (i >= (long)BB * XCATC) return;
    int b = (int)(i / XCATC), c = (int)(i % XCATC);
    float mx = -1e30f, s = 0.0f;
    for (int g = 0; g < GG; ++g) {
        float v = xcat[((long)b * GG + g) * XCATC + c];
        mx = fmaxf(mx, v); s += v;
    }
    glob[(long)b * 2 * XCATC + c]          = mx;
    glob[(long)b * 2 * XCATC + XCATC + c]  = s / GG;
}

__global__ void knn3_k(const float* __restrict__ xyz, const float* __restrict__ center,
                       int* __restrict__ idx3, float* __restrict__ w3) {
    long i = (long)blockIdx.x * blockDim.x + threadIdx.x;   // over BN
    if (i >= (long)BN) return;
    int b = (int)(i / NP);
    float px = xyz[i * 3], py = xyz[i * 3 + 1], pz = xyz[i * 3 + 2];
    float d0 = 1e30f, d1 = 1e30f, d2 = 1e30f;
    int   i0 = 0, i1 = 0, i2 = 0;
    for (int s = 0; s < GG; ++s) {
        const float* c = center + ((long)b * GG + s) * 3;
        float dx = px - c[0], dy = py - c[1], dz = pz - c[2];
        float d = dx * dx + dy * dy + dz * dz;
        if (d < d0)      { d2 = d1; i2 = i1; d1 = d0; i1 = i0; d0 = d; i0 = s; }
        else if (d < d1) { d2 = d1; i2 = i1; d1 = d;  i1 = s; }
        else if (d < d2) { d2 = d;  i2 = s; }
    }
    float r0 = 1.0f / (sqrtf(fmaxf(d0, 0.0f)) + 1e-8f);
    float r1 = 1.0f / (sqrtf(fmaxf(d1, 0.0f)) + 1e-8f);
    float r2 = 1.0f / (sqrtf(fmaxf(d2, 0.0f)) + 1e-8f);
    float s  = r0 + r1 + r2;
    idx3[i * 3 + 0] = i0; idx3[i * 3 + 1] = i1; idx3[i * 3 + 2] = i2;
    w3[i * 3 + 0] = r0 / s; w3[i * 3 + 1] = r1 / s; w3[i * 3 + 2] = r2 / s;
}

// builds [xyz | interp] rows padded to PKIN cols (zeros beyond 1155)
__global__ void interp_concat_k(const float* __restrict__ xyz, const float* __restrict__ xcat,
                                const int* __restrict__ idx3, const float* __restrict__ w3,
                                float* __restrict__ out) {
    long i = (long)blockIdx.x * blockDim.x + threadIdx.x;   // over BN*PKIN
    if (i >= (long)BN * PKIN) return;
    long row = i / PKIN; int c = (int)(i % PKIN);
    if (c >= 1155) { out[i] = 0.0f; return; }
    if (c < 3)     { out[i] = xyz[row * 3 + c]; return; }
    int cc = c - 3;
    int b = (int)(row / NP);
    long base = row * 3;
    float v = 0.0f;
    for (int j = 0; j < 3; ++j)
        v += w3[base + j] * xcat[((long)b * GG + idx3[base + j]) * XCATC + cc];
    out[i] = v;
}

__global__ void head_concat_k(const float* __restrict__ pf2, const float* __restrict__ glob,
                              float* __restrict__ out) {
    long i = (long)blockIdx.x * blockDim.x + threadIdx.x;   // over BN*3328
    if (i >= (long)BN * 3328) return;
    long row = i / 3328; int c = (int)(i % 3328);
    if (c < 1024) out[i] = pf2[row * 1024 + c];
    else {
        int b = (int)(row / NP);
        out[i] = glob[(long)b * 2304 + (c - 1024)];
    }
}

__global__ __launch_bounds__(128)
void logsoftmax_inplace_k(float* __restrict__ x, long rows, int D) {
    long row = (long)blockIdx.x * 4 + (threadIdx.x >> 5);
    int lane = threadIdx.x & 31;
    if (row >= rows) return;
    float* p = x + row * D;
    float mx = -1e30f;
    for (int c = lane; c < D; c += 32) mx = fmaxf(mx, p[c]);
    for (int o = 16; o > 0; o >>= 1) mx = fmaxf(mx, __shfl_xor(mx, o, 32));
    float s = 0.0f;
    for (int c = lane; c < D; c += 32) s += expf(p[c] - mx);
    for (int o = 16; o > 0; o >>= 1) s += __shfl_xor(s, o, 32);
    float lse = mx + logf(s);
    for (int c = lane; c < D; c += 32) p[c] -= lse;
}

// ---------------------------------------------------------------------------
// Host orchestration
// ---------------------------------------------------------------------------
extern "C" void kernel_launch(void* const* d_in, const int* in_sizes, int n_in,
                              void* d_out, int out_size, void* d_ws, size_t ws_size,
                              hipStream_t stream)
{
    (void)in_sizes; (void)n_in; (void)out_size; (void)ws_size;

    // --- input pointer map (pts first, then params in jax pytree order:
    //     sorted dict keys, lists in order; lin={b,w}, bn={b,g,m,v}, ln={b,g}) ---
    const float* pts = (const float*)d_in[0];
    auto P = [&](int i) -> const float* { return (const float*)d_in[i]; };
    auto BPx = [&](int blk, int j) { return 1 + 11 * blk + j; };
    // block j: 0 fc1.b 1 fc1.w 2 fc2.b 3 fc2.w 4 ln1.b 5 ln1.g 6 ln2.b 7 ln2.g
    //          8 proj.b 9 proj.w 10 wqkv
    const int E_BN1 = 133, E_BN3 = 137, E_C1B = 141, E_C1W = 142, E_C2B = 143,
              E_C2W = 144, E_C3B = 145, E_C3W = 146, E_C4B = 147, E_C4W = 148;
    const int H_BN1 = 149, H_BN2 = 153, H_C1B = 157, H_C1W = 158, H_C2B = 159,
              H_C2W = 160, H_C3B = 161, H_C3W = 162;
    const int NORM_B = 163, NORM_G = 164;
    const int POS_L1B = 165, POS_L1W = 166, POS_L2B = 167, POS_L2W = 168;
    const int P_BN1 = 169, P_BN2 = 173, P_C1B = 177, P_C1W = 178, P_C2B = 179,
              P_C2W = 180;

    // --- workspace bump allocator ---
    char* wsb = (char*)d_ws;
    size_t off = 0;
    auto allocB = [&](size_t bytes) -> void* {
        off = (off + 255) & ~(size_t)255;
        void* p = wsb + off;
        off += bytes;
        return p;
    };
    auto allocF = [&](long n) { return (float*)allocB((size_t)n * 4); };
    auto allocI = [&](long n) { return (int*)allocB((size_t)n * 4); };
    auto allocH16 = [&](long n) { return (f16_t*)allocB((size_t)n * 2); };

    auto pack = [&](const float* src, f16_t* dst, long n) {
        pack_f16_k<<<cdiv(n, 256), 256, 0, stream>>>(src, dst, n);
    };
    // fp32 [K,N] -> f16 [N,Kpad] transposed
    auto packT = [&](const float* src, f16_t* dst, int K, int N, int Kpad) {
        packT_k<<<cdiv((long)N * Kpad, 256), 256, 0, stream>>>(src, dst, K, N, Kpad);
    };

    // GEMM launcher: W is f16 [N,K] with row stride ldw
    auto gemm = [&](const float* A, int lda, long sAb, long sAh,
                    const f16_t* W, int ldw, long sWb, long sWh,
                    const float* bias, const float* resid,
                    float* C, int ldc, long sCb, long sCh,
                    int M, int N, int K, int batB, int batH, float alpha, int act,
                    const float* bng = nullptr, const float* bnb2 = nullptr,
                    const float* bnm = nullptr, const float* bnv = nullptr) {
        dim3 grid(cdiv(N, 64), cdiv(M, 64), batB * batH);
        gemm_wmma<<<grid, 128, 0, stream>>>(A, lda, sAb, sAh, W, ldw, sWb, sWh,
                                            bias, bng, bnb2, bnm, bnv, resid,
                                            C, ldc, sCb, sCh, M, N, K, batH, alpha, act);
    };

    // --- persistent buffers ---
    float* xyz    = allocF((long)BB * NP * 3);
    float* center = allocF((long)BB * GG * 3);
    int*   fpsI   = allocI((long)BB * GG);
    int*   nnI    = allocI((long)BB * GG * KG);
    float* tokens = allocF((long)BT * DD);
    float* posb   = allocF((long)BT * DD);
    float* xbuf   = allocF((long)BT * DD);
    float* xcat   = allocF((long)BT * XCATC);
    float* glob   = allocF((long)BB * 2 * XCATC);
    int*   idx3   = allocI((long)BN * 3);
    float* w3     = allocF((long)BN * 3);
    // transformer scratch (reused across all 12 blocks)
    float* xin    = allocF((long)BT * DD);
    float* ln1o   = allocF((long)BT * DD);
    float* qkv    = allocF((long)BT * 1152);
    f16_t* qkvh   = allocH16((long)BT * 1152);
    f16_t* vt     = allocH16((long)BB * NH * DH * TT);
    float* scores = allocF((long)BB * NH * TT * TT);
    float* attnO  = allocF((long)BT * DD);
    float* x1     = allocF((long)BT * DD);
    float* ln2o   = allocF((long)BT * DD);
    float* mh1    = allocF((long)BT * 1536);
    float* pe     = allocF((long)BT * DD);
    float* pt1    = allocF((long)BT * DD);

    // --- f16 weights, packed transposed [N,K] (persistent) ---
    f16_t* h_c1 = allocH16(128 * 32);       packT(P(E_C1W), h_c1, 3, 128, 32);
    f16_t* h_c2 = allocH16(256 * 128);      packT(P(E_C2W), h_c2, 128, 256, 128);
    f16_t* h_c3 = allocH16(512 * 512);      packT(P(E_C3W), h_c3, 512, 512, 512);
    f16_t* h_c4 = allocH16(384 * 512);      packT(P(E_C4W), h_c4, 512, 384, 512);
    f16_t* h_l1 = allocH16(384 * 384);      packT(P(POS_L1W), h_l1, 384, 384, 384);
    f16_t* h_l2 = allocH16(384 * 384);      packT(P(POS_L2W), h_l2, 384, 384, 384);
    f16_t *h_qkv[12], *h_proj[12], *h_fc1[12], *h_fc2[12];
    for (int i = 0; i < 12; ++i) {
        h_qkv[i]  = allocH16(1152 * 384);   packT(P(BPx(i, 10)), h_qkv[i], 384, 1152, 384);
        h_proj[i] = allocH16(384 * 384);    packT(P(BPx(i, 9)),  h_proj[i], 384, 384, 384);
        h_fc1[i]  = allocH16(1536 * 384);   packT(P(BPx(i, 1)),  h_fc1[i], 384, 1536, 384);
        h_fc2[i]  = allocH16(384 * 1536);   packT(P(BPx(i, 3)),  h_fc2[i], 1536, 384, 1536);
    }
    f16_t* h_p1 = allocH16(1536L * PKIN);   packT(P(P_C1W), h_p1, 1155, 1536, PKIN);
    f16_t* h_p2 = allocH16(1024L * 1536);   packT(P(P_C2W), h_p2, 1536, 1024, 1536);
    f16_t* h_h1 = allocH16(512L * 3328);    packT(P(H_C1W), h_h1, 3328, 512, 3328);
    f16_t* h_h2 = allocH16(256 * 512);      packT(P(H_C2W), h_h2, 512, 256, 512);
    f16_t* h_h3 = allocH16(CLS * 256);      packT(P(H_C3W), h_h3, 256, CLS, 256);

    const size_t mark = off;   // start of phase-scratch region

    // ------------------------------------------------------------------
    // Stage 1: grouping
    // ------------------------------------------------------------------
    transpose_pts_k<<<cdiv((long)BB * NP, 256), 256, 0, stream>>>(pts, xyz);
    fps_k<<<BB, 256, 0, stream>>>(xyz, fpsI);
    gather_center_k<<<cdiv((long)BB * GG, 256), 256, 0, stream>>>(xyz, fpsI, center);
    knn_k<<<BB * GG, 256, 0, stream>>>(xyz, center, nnI);

    // ------------------------------------------------------------------
    // Stage 2: encoder (scratch phase A)
    // ------------------------------------------------------------------
    {
        float* nbr  = allocF((long)BGK * 32);   // padded K
        float* f1   = allocF((long)BGK * 128);
        float* f2   = allocF((long)BGK * 256);
        float* g256 = allocF((long)BB * GG * 256);
        float* f3in = allocF((long)BGK * 512);
        float* f3   = allocF((long)BGK * 512);
        float* f4   = allocF((long)BGK * 384);

        build_nbr_k<<<cdiv((long)BGK, 256), 256, 0, stream>>>(xyz, center, nnI, nbr);
        // c1 + bn1 + relu (K padded 3->32)
        gemm(nbr, 32, 0, 0, h_c1, 32, 0, 0, P(E_C1B), nullptr,
             f1, 128, 0, 0, BGK, 128, 32, 1, 1, 1.0f, 1,
             P(E_BN1 + 1), P(E_BN1 + 0), P(E_BN1 + 2), P(E_BN1 + 3));
        // c2
        gemm(f1, 128, 0, 0, h_c2, 128, 0, 0, P(E_C2B), nullptr,
             f2, 256, 0, 0, BGK, 256, 128, 1, 1, 1.0f, 0);
        pool_max_k_k<<<cdiv((long)BB * GG * 256, 256), 256, 0, stream>>>(f2, g256, BB * GG, KG, 256);
        concat512_k<<<cdiv((long)BGK * 512, 256), 256, 0, stream>>>(g256, f2, f3in);
        // c3 + bn3 + relu
        gemm(f3in, 512, 0, 0, h_c3, 512, 0, 0, P(E_C3B), nullptr,
             f3, 512, 0, 0, BGK, 512, 512, 1, 1, 1.0f, 1,
             P(E_BN3 + 1), P(E_BN3 + 0), P(E_BN3 + 2), P(E_BN3 + 3));
        // c4
        gemm(f3, 512, 0, 0, h_c4, 512, 0, 0, P(E_C4B), nullptr,
             f4, 384, 0, 0, BGK, 384, 512, 1, 1, 1.0f, 0);
        pool_max_k_k<<<cdiv((long)BB * GG * 384, 256), 256, 0, stream>>>(f4, tokens, BB * GG, KG, 384);
    }
    off = mark;   // release encoder scratch

    // ------------------------------------------------------------------
    // Stage 3: positional embedding
    // ------------------------------------------------------------------
    pos_embed_k<<<cdiv((long)BT * DD, 256), 256, 0, stream>>>(center, pe);
    gemm(pe, DD, 0, 0, h_l1, DD, 0, 0, P(POS_L1B), nullptr,
         pt1, DD, 0, 0, BT, DD, DD, 1, 1, 1.0f, 2);
    gemm(pt1, DD, 0, 0, h_l2, DD, 0, 0, P(POS_L2B), nullptr,
         posb, DD, 0, 0, BT, DD, DD, 1, 1, 1.0f, 0);

    // ------------------------------------------------------------------
    // Stage 4: 12 transformer blocks
    // ------------------------------------------------------------------
    int seg = 0;
    for (int i = 0; i < 12; ++i) {
        const float* xsrc = (i == 0) ? tokens : xbuf;
        add_vec_k<<<cdiv((long)BT * DD, 256), 256, 0, stream>>>(xsrc, posb, xin, (long)BT * DD);
        layernorm_k<<<cdiv(BT, 4), 128, 0, stream>>>(xin, DD, P(BPx(i, 5)), P(BPx(i, 4)),
                                                     ln1o, DD, 0, BT, DD);
        // qkv
        gemm(ln1o, DD, 0, 0, h_qkv[i], DD, 0, 0, nullptr, nullptr,
             qkv, 1152, 0, 0, BT, 1152, DD, 1, 1, 1.0f, 0);
        pack(qkv, qkvh, (long)BT * 1152);
        pack_vT_k<<<cdiv((long)BB * NH * DH * TT, 256), 256, 0, stream>>>(qkv, vt);
        // scores = Q K^T * 1/8  (batched over B x H); K rows contiguous in qkvh
        gemm(qkv, 1152, (long)TT * 1152, 64,
             qkvh + 384, 1152, (long)TT * 1152, 64,
             nullptr, nullptr,
             scores, TT, (long)NH * TT * TT, (long)TT * TT,
             TT, TT, DH, BB, NH, 0.125f, 0);
        softmax_rows_k<<<cdiv((long)BB * NH * TT, 4), 128, 0, stream>>>(scores, (long)BB * NH * TT, TT);
        // O = att V  (V transposed pack, contiguous along token dim)
        gemm(scores, TT, (long)NH * TT * TT, (long)TT * TT,
             vt, TT, (long)NH * DH * TT, (long)DH * TT,
             nullptr, nullptr,
             attnO, DD, (long)TT * DD, 64,
             TT, DH, TT, BB, NH, 1.0f, 0);
        // proj (+ residual xin)
        gemm(attnO, DD, 0, 0, h_proj[i], DD, 0, 0, P(BPx(i, 8)), xin,
             x1, DD, 0, 0, BT, DD, DD, 1, 1, 1.0f, 0);
        layernorm_k<<<cdiv(BT, 4), 128, 0, stream>>>(x1, DD, P(BPx(i, 7)), P(BPx(i, 6)),
                                                     ln2o, DD, 0, BT, DD);
        // fc1 + gelu
        gemm(ln2o, DD, 0, 0, h_fc1[i], DD, 0, 0, P(BPx(i, 0)), nullptr,
             mh1, 1536, 0, 0, BT, 1536, DD, 1, 1, 1.0f, 2);
        // fc2 (+ residual x1)
        gemm(mh1, 1536, 0, 0, h_fc2[i], 1536, 0, 0, P(BPx(i, 2)), x1,
             xbuf, DD, 0, 0, BT, DD, 1536, 1, 1, 1.0f, 0);
        if (i == 3 || i == 7 || i == 11) {
            layernorm_k<<<cdiv(BT, 4), 128, 0, stream>>>(xbuf, DD, P(NORM_G), P(NORM_B),
                                                         xcat, XCATC, seg * DD, BT, DD);
            ++seg;
        }
    }

    // ------------------------------------------------------------------
    // Stage 5: global pooling + 3-NN interpolation (scratch phase B)
    // ------------------------------------------------------------------
    pool_global_k<<<cdiv((long)BB * XCATC, 256), 256, 0, stream>>>(xcat, glob);
    knn3_k<<<cdiv((long)BN, 256), 256, 0, stream>>>(xyz, center, idx3, w3);

    const size_t markB = off;
    float* pf_in = allocF((long)BN * PKIN);
    float* pf1   = allocF((long)BN * 1536);
    interp_concat_k<<<cdiv((long)BN * PKIN, 256), 256, 0, stream>>>(xyz, xcat, idx3, w3, pf_in);
    // prop c1 + bn1 + relu (K padded 1155->1184)
    gemm(pf_in, PKIN, 0, 0, h_p1, PKIN, 0, 0, P(P_C1B), nullptr,
         pf1, 1536, 0, 0, BN, 1536, PKIN, 1, 1, 1.0f, 1,
         P(P_BN1 + 1), P(P_BN1 + 0), P(P_BN1 + 2), P(P_BN1 + 3));
    // prop c2 + bn2 + relu  (pf2 reuses pf_in region: 1024 <= PKIN cols)
    float* pf2 = pf_in;
    gemm(pf1, 1536, 0, 0, h_p2, 1536, 0, 0, P(P_C2B), nullptr,
         pf2, 1024, 0, 0, BN, 1024, 1536, 1, 1, 1.0f, 1,
         P(P_BN2 + 1), P(P_BN2 + 0), P(P_BN2 + 2), P(P_BN2 + 3));

    // head scratch reuses pf1's region
    off = markB;
    (void)allocF((long)BN * PKIN);               // keep pf2 region reserved
    float* hin = allocF((long)BN * 3328);
    float* hh1 = allocF((long)BN * 512);
    float* hh2 = allocF((long)BN * 256);

    head_concat_k<<<cdiv((long)BN * 3328, 256), 256, 0, stream>>>(pf2, glob, hin);
    gemm(hin, 3328, 0, 0, h_h1, 3328, 0, 0, P(H_C1B), nullptr,
         hh1, 512, 0, 0, BN, 512, 3328, 1, 1, 1.0f, 1,
         P(H_BN1 + 1), P(H_BN1 + 0), P(H_BN1 + 2), P(H_BN1 + 3));
    gemm(hh1, 512, 0, 0, h_h2, 512, 0, 0, P(H_C2B), nullptr,
         hh2, 256, 0, 0, BN, 256, 512, 1, 1, 1.0f, 1,
         P(H_BN2 + 1), P(H_BN2 + 0), P(H_BN2 + 2), P(H_BN2 + 3));
    // final classifier straight into d_out, then in-place log-softmax
    float* logits = (float*)d_out;
    gemm(hh2, 256, 0, 0, h_h3, 256, 0, 0, P(H_C3B), nullptr,
         logits, CLS, 0, 0, BN, CLS, 256, 1, 1, 1.0f, 0);
    logsoftmax_inplace_k<<<cdiv((long)BN, 4), 128, 0, stream>>>(logits, BN, CLS);
}
